// TransFusionHeadGaussianMatching_79723182948965
// MI455X (gfx1250) — matrix operand order repaired and checked by reading, not compiled, not run on hardware
//
#include <hip/hip_runtime.h>
#include <hip/hip_bf16.h>

// ---------------- problem constants ----------------
#define Bq    2
#define CIN0  512
#define Hd    180
#define Wd    180
#define HWp   (Hd*Wd)      // 32400
#define CHN   128
#define NPq   200
#define NHh   8
#define DH    16
#define FFNh  256
#define NCc   10
#define NSEL  (NCc*HWp)    // 324000
#define OUTR  13
#define BN_S  0.99999500003749968750f
#define SQRT3 1.7320508075688772f

typedef __attribute__((ext_vector_type(16))) __bf16 bf16x16;
typedef __attribute__((ext_vector_type(8)))  float  v8f;

__constant__ float c_anch[NCc][3] = {
  {4.63f,1.97f,1.74f},{6.93f,2.51f,2.84f},{6.37f,2.85f,3.19f},{10.5f,2.94f,3.47f},
  {12.29f,2.9f,3.87f},{0.5f,2.53f,0.98f},{2.11f,0.77f,1.47f},{1.7f,0.6f,1.28f},
  {0.73f,0.67f,1.77f},{0.41f,0.41f,1.07f}};
__constant__ int c_rowhead[OUTR] = {0,0,1,2,2,2,3,3,4,4,5,6,7};

// ---------------- weight prep: W[O][I][KH][KW] f32 -> WT[KHW][I][128] bf16 ---
__global__ void k_prep_w(const float* __restrict__ W, __bf16* __restrict__ WT,
                         int I, int KHW) {
  int idx = blockIdx.x*256 + threadIdx.x;
  int total = KHW*I*CHN;
  if (idx >= total) return;
  int khw = idx / (I*CHN);
  int rem = idx - khw*I*CHN;
  int i   = rem / CHN;
  int o   = rem - i*CHN;
  WT[idx] = (__bf16)W[((size_t)o*I + i)*KHW + khw];
}

// ---------------- input transpose+cast: [B][C][HW] f32 -> [B][HW][C] bf16 ----
__global__ void k_castT(const float* __restrict__ in, __bf16* __restrict__ out) {
  size_t idx = (size_t)blockIdx.x*256 + threadIdx.x;
  if (idx >= (size_t)Bq*CIN0*HWp) return;
  int b = (int)(idx / ((size_t)CIN0*HWp));
  size_t rem = idx - (size_t)b*CIN0*HWp;
  int c = (int)(rem / HWp);
  int p = (int)(rem - (size_t)c*HWp);
  out[((size_t)b*HWp + p)*CIN0 + c] = (__bf16)in[idx];
}

// ---------------- implicit-GEMM conv / 1x1-GEMM via WMMA bf16 ---------------
// inT: [B][HW][CIN] bf16 (channel-last), WT: [KHW][CIN][128] bf16
// mode 0: out = acc + bias(p0f);  mode 1: out = relu(acc*(p0f*BN_S)+p1f)
__global__ __launch_bounds__(256)
void k_conv_wmma(const __bf16* __restrict__ inT, const __bf16* __restrict__ WT,
                 const float* __restrict__ p0f, const float* __restrict__ p1f,
                 float* __restrict__ outF, __bf16* __restrict__ outB,
                 int CIN, int KHW, int mode) {
  int tid  = threadIdx.x;
  int wave = tid >> 5, lane = tid & 31, lm = lane & 15, lh = lane >> 4;
  int b    = blockIdx.z;
  int pbase = blockIdx.x * 128;

  __shared__ alignas(32) __bf16 shA[256][16];      // 8 m-tiles x 32 lanes x 16
  __shared__ alignas(32) __bf16 shB[8][32][16];    // 8 n-tiles x 32 lanes x 16

  v8f acc[8];
  v8f zz = {0.f,0.f,0.f,0.f,0.f,0.f,0.f,0.f};
#pragma unroll
  for (int t=0;t<8;++t) acc[t]=zz;

  // A staging slot decode (constant across K loop)
  int amt = tid >> 5, asl = tid & 31;
  int am  = amt*16 + (asl & 15);
  int ah  = asl >> 4;
  int ap  = pbase + am;
  int ay  = ap / Wd, ax = ap - ay*Wd;
  // B staging slot decode
  int br = tid >> 3, bpart = tid & 7;
  int blb = (br >> 4)*16, be = br & 15;

  for (int khw = 0; khw < KHW; ++khw) {
    int dy = 0, dx = 0;
    if (KHW == 9) { dy = khw/3 - 1; dx = khw - (khw/3)*3 - 1; }
    int sy = ay + dy, sx = ax + dx;
    bool avalid;
    long aq;
    if (KHW == 1) { avalid = (ap < HWp); aq = ap; }
    else { avalid = (ap < HWp) && (sy >= 0) && (sy < Hd) && (sx >= 0) && (sx < Wd);
           aq = (long)sy*Wd + sx; }
    const __bf16* abase = inT + ((size_t)b*HWp + (avalid ? aq : 0))*CIN;

    for (int c0 = 0; c0 < CIN; c0 += 32) {
      __syncthreads();
      { // stage A (fragment layout): lane holds K = {ah*16..+15}
        uint4 v0 = make_uint4(0,0,0,0), v1 = make_uint4(0,0,0,0);
        if (avalid) {
          const uint4* src = (const uint4*)(abase + c0 + ah*16);
          v0 = src[0]; v1 = src[1];
        }
        uint4* d = (uint4*)&shA[tid][0];
        d[0] = v0; d[1] = v1;
      }
      { // stage B: coalesced 32B global read of one K-row chunk, scatter to frag
        const __bf16* src = WT + ((size_t)khw*CIN + c0 + br)*CHN + bpart*16;
        uint4 w0 = ((const uint4*)src)[0];
        uint4 w1 = ((const uint4*)src)[1];
        __bf16 tmp[16];
        *(uint4*)&tmp[0] = w0; *(uint4*)&tmp[8] = w1;
#pragma unroll
        for (int j=0;j<16;++j) shB[bpart][blb + j][be] = tmp[j];
      }
      __syncthreads();
      bf16x16 af = *(const bf16x16*)&shA[wave*32 + lane][0];
#pragma unroll
      for (int t=0;t<8;++t) {
        bf16x16 bfm = *(const bf16x16*)&shB[t][lane][0];
        acc[t] = __builtin_amdgcn_wmma_f32_16x16x32_bf16(
                    false, af, false, bfm, (short)0, acc[t], false, false);
      }
    }
  }

  // epilogue: D layout lane l -> N=l&15, M = v + (l>>4)*8
#pragma unroll
  for (int t=0;t<8;++t) {
    int n = t*16 + lm;
    float bi = 0.f, gg = 0.f, bb = 0.f;
    if (mode == 0 && p0f) bi = p0f[n];
    if (mode == 1) { gg = p0f[n]*BN_S; bb = p1f[n]; }
#pragma unroll
    for (int v=0;v<8;++v) {
      int p = pbase + wave*16 + lh*8 + v;
      if (p < HWp) {
        float val = acc[t][v];
        if (mode == 0) val += bi;
        else if (mode == 1) val = fmaxf(val*gg + bb, 0.f);
        size_t o = ((size_t)b*HWp + p)*CHN + n;
        if (outF) outF[o] = val;
        if (outB) outB[o] = (__bf16)val;
      }
    }
  }
}

// ---------------- 128->1 3x3 heatmap conv (tiny) ----------------------------
__global__ void k_hm(const __bf16* __restrict__ t, const float* __restrict__ whm,
                     const float* __restrict__ bhm, float* __restrict__ hm1) {
  __shared__ float sw[CHN*9];
  int tid = threadIdx.x;
  for (int i = tid; i < CHN*9; i += blockDim.x) sw[i] = whm[i];
  __syncthreads();
  int idx = blockIdx.x*blockDim.x + tid;
  if (idx >= Bq*HWp) return;
  int b = idx / HWp, p = idx - b*HWp;
  int y = p / Wd, x = p - y*Wd;
  float acc = bhm[0];
  for (int kh=0; kh<3; ++kh) {
    int sy = y + kh - 1; if (sy < 0 || sy >= Hd) continue;
    for (int kw=0; kw<3; ++kw) {
      int sx = x + kw - 1; if (sx < 0 || sx >= Wd) continue;
      const __bf16* tr = t + ((size_t)b*HWp + (size_t)sy*Wd + sx)*CHN;
      for (int ci=0; ci<CHN; ++ci) acc += (float)tr[ci] * sw[ci*9 + kh*3 + kw];
    }
  }
  hm1[idx] = acc;
}

// ---------------- anchors / small precompute --------------------------------
__global__ void k_small_pre(const float* w1, const float* b1, const float* w2,
                            const float* b2, const float* aq_w,
                            const float* hm_scale, const float* hm_bias,
                            float* sK, float* aqc, float* scal) {
  int t = threadIdx.x;
  if (t < NCc) {
    float a0 = logf(c_anch[t][0]), a1 = logf(c_anch[t][1]), a2 = logf(c_anch[t][2]);
    float h[3], av[3];
    for (int i=0;i<3;++i)
      h[i] = fmaxf(0.f, w1[i*3+0]*a0 + w1[i*3+1]*a1 + w1[i*3+2]*a2 + b1[i]);
    for (int o=0;o<3;++o)
      av[o] = w2[o*3+0]*h[0] + w2[o*3+1]*h[1] + w2[o*3+2]*h[2] + b2[o];
    float nrm = sqrtf(av[0]*av[0] + av[1]*av[1] + av[2]*av[2]);
    sK[t] = (av[0]+av[1]+av[2]) / (1e-8f + nrm);
  }
  if (t < CHN) {
    for (int k=0;k<NCc;++k) {
      float a0 = logf(c_anch[k][0]), a1 = logf(c_anch[k][1]), a2 = logf(c_anch[k][2]);
      aqc[k*CHN + t] = aq_w[t*3+0]*a0 + aq_w[t*3+1]*a1 + aq_w[t*3+2]*a2;
    }
  }
  if (t == 0) { scal[0] = expf(hm_scale[0]); scal[1] = hm_bias[0]; }
}

// ---------------- heatmap (analytic collapse of the 3-row norm) -------------
__global__ void k_heat(const float* __restrict__ hm1, const float* __restrict__ sK,
                       const float* __restrict__ scal, float* __restrict__ heat) {
  int idx = blockIdx.x*256 + threadIdx.x;
  if (idx >= Bq*NCc*HWp) return;
  int b = idx / (NCc*HWp);
  int rem = idx - b*NCc*HWp;
  int k = rem / HWp, p = rem - k*HWp;
  float d  = hm1[b*HWp + p];
  float dn = d / (1e-8f + SQRT3*fabsf(d));
  float z  = scal[0]*sK[k]*dn + scal[1];
  heat[idx] = 1.f/(1.f + expf(-z));
}

// ---------------- 3x3 max-pool NMS ------------------------------------------
__global__ void k_nms(const float* __restrict__ heat, float* __restrict__ nms) {
  int idx = blockIdx.x*256 + threadIdx.x;
  if (idx >= Bq*NCc*HWp) return;
  int b = idx / (NCc*HWp);
  int rem = idx - b*NCc*HWp;
  int k = rem / HWp, p = rem - k*HWp;
  int y = p / Wd, x = p - y*Wd;
  const float* base = heat + ((size_t)b*NCc + k)*HWp;
  float c = base[p];
  float m = -3.4e38f;
  for (int dy=-1; dy<=1; ++dy) {
    int ny = y + dy; if (ny < 0 || ny >= Hd) continue;
    for (int dx=-1; dx<=1; ++dx) {
      int nx = x + dx; if (nx < 0 || nx >= Wd) continue;
      m = fmaxf(m, base[ny*Wd + nx]);
    }
  }
  nms[idx] = (c == m) ? c : 0.f;
}

// ---------------- exact iterative top-200 (tie: lowest index) ---------------
__global__ __launch_bounds__(1024)
void k_topk(float* __restrict__ nms, int* __restrict__ cls, int* __restrict__ sp) {
  int b = blockIdx.x, tid = threadIdx.x;
  float* base = nms + (size_t)b*NSEL;
  __shared__ float sv[1024];
  __shared__ int   si[1024];
  for (int it = 0; it < NPq; ++it) {
    float bv = -3.4e38f; int bi = NSEL;
    for (int i = tid; i < NSEL; i += 1024) {
      float v = base[i];
      if (v > bv || (v == bv && i < bi)) { bv = v; bi = i; }
    }
    sv[tid] = bv; si[tid] = bi; __syncthreads();
    for (int off = 512; off > 0; off >>= 1) {
      if (tid < off) {
        float v2 = sv[tid+off]; int i2 = si[tid+off];
        if (v2 > sv[tid] || (v2 == sv[tid] && i2 < si[tid])) { sv[tid]=v2; si[tid]=i2; }
      }
      __syncthreads();
    }
    if (tid == 0) {
      int bi0 = si[0];
      cls[b*NPq + it] = bi0 / HWp;
      sp [b*NPq + it] = bi0 - (bi0/HWp)*HWp;
      base[bi0] = -3.4e38f;
    }
    __syncthreads();
  }
}

// ---------------- build queries: gather + anchor feat + query pos-embed -----
__global__ void k_buildq(const int* __restrict__ sp, const int* __restrict__ cls,
                         const float* __restrict__ lidarF, const float* __restrict__ aqc,
                         const float* __restrict__ aq_b,
                         const float* pw1, const float* pb1, const float* pg,
                         const float* pbt, const float* pw2, const float* pb2,
                         float* __restrict__ qf, float* __restrict__ qpe) {
  int q = blockIdx.x, b = blockIdx.y, c = threadIdx.x;
  int s = sp[b*NPq + q], cl = cls[b*NPq + q];
  float px = (float)(s / Wd) + 0.5f;
  float py = (float)(s - (s / Wd)*Wd) + 0.5f;
  __shared__ float h[CHN];
  float hv = pw1[c*2+0]*px + pw1[c*2+1]*py + pb1[c];
  hv = hv*(pg[c]*BN_S) + pbt[c];
  h[c] = fmaxf(hv, 0.f);
  float qv = lidarF[((size_t)b*HWp + s)*CHN + c] + aqc[cl*CHN + c] + aq_b[c];
  __syncthreads();
  float acc = pb2[c];
  for (int k=0; k<CHN; ++k) acc += pw2[c*CHN + k]*h[k];
  size_t o = ((size_t)b*NPq + q)*CHN + c;
  qf[o] = qv; qpe[o] = acc;
}

// ---------------- key pos-embed (batch-invariant) fused into K/V input ------
__global__ void k_kpe(const float* __restrict__ lidarF,
                      const float* pw1, const float* pb1, const float* pg,
                      const float* pbt, const float* pw2, const float* pb2,
                      __bf16* __restrict__ kv) {
  int p = blockIdx.x, c = threadIdx.x;
  float px = (float)(p / Wd) + 0.5f;
  float py = (float)(p - (p / Wd)*Wd) + 0.5f;
  __shared__ float h[CHN];
  float hv = pw1[c*2+0]*px + pw1[c*2+1]*py + pb1[c];
  hv = hv*(pg[c]*BN_S) + pbt[c];
  h[c] = fmaxf(hv, 0.f);
  __syncthreads();
  float acc = pb2[c];
  for (int k=0; k<CHN; ++k) acc += pw2[c*CHN + k]*h[k];
  for (int b=0; b<Bq; ++b) {
    size_t o = ((size_t)b*HWp + p)*CHN + c;
    kv[o] = (__bf16)(lidarF[o] + acc);
  }
}

// ---------------- small GEMM: C[b][m][N] = (A(+A2))[b][m][:K] @ W[N][K]^T ---
__global__ void k_gemm_small(const float* __restrict__ A, const float* __restrict__ A2,
                             const float* __restrict__ W, const float* __restrict__ bias,
                             float* __restrict__ C, int K, int N, int relu) {
  int m = blockIdx.x, b = blockIdx.y, n = threadIdx.x;
  extern __shared__ float sA[];
  const float* ar  = A  + ((size_t)b*NPq + m)*K;
  const float* ar2 = A2 ? A2 + ((size_t)b*NPq + m)*K : nullptr;
  for (int i = n; i < K; i += blockDim.x) sA[i] = ar[i] + (ar2 ? ar2[i] : 0.f);
  __syncthreads();
  float acc = bias[n];
  const float* wr = W + (size_t)n*K;
  for (int k=0; k<K; ++k) acc += sA[k]*wr[k];
  if (relu) acc = fmaxf(acc, 0.f);
  C[((size_t)b*NPq + m)*N + n] = acc;
}

// ---------------- self-attention over 200 queries ---------------------------
__global__ __launch_bounds__(256)
void k_attn_small(const float* __restrict__ Q, const float* __restrict__ K,
                  const float* __restrict__ V, float* __restrict__ O) {
  int qi = blockIdx.x, b = blockIdx.y, tid = threadIdx.x;
  __shared__ float sq[CHN];
  __shared__ float ss[NHh*NPq];
  if (tid < CHN) sq[tid] = Q[((size_t)b*NPq + qi)*CHN + tid];
  __syncthreads();
  for (int idx = tid; idx < NHh*NPq; idx += 256) {
    int h = idx / NPq, k = idx - h*NPq;
    const float* kr = K + ((size_t)b*NPq + k)*CHN + h*DH;
    float s = 0.f;
    for (int d=0; d<DH; ++d) s += sq[h*DH + d]*kr[d];
    ss[idx] = s*0.25f;
  }
  __syncthreads();
  if (tid < NHh) {
    float m = -3.4e38f;
    for (int k=0;k<NPq;++k) m = fmaxf(m, ss[tid*NPq + k]);
    float sum = 0.f;
    for (int k=0;k<NPq;++k) { float e = expf(ss[tid*NPq + k] - m); ss[tid*NPq + k] = e; sum += e; }
    float inv = 1.f/sum;
    for (int k=0;k<NPq;++k) ss[tid*NPq + k] *= inv;
  }
  __syncthreads();
  if (tid < CHN) {
    int h = tid / DH;
    float acc = 0.f;
    for (int k=0;k<NPq;++k) acc += ss[h*NPq + k]*V[((size_t)b*NPq + k)*CHN + tid];
    O[((size_t)b*NPq + qi)*CHN + tid] = acc;
  }
}

// ---------------- flash cross-attention: 200 q x 32400 keys -----------------
__global__ __launch_bounds__(256)
void k_flash(const float* __restrict__ Q, const float* __restrict__ K,
             const float* __restrict__ V, float* __restrict__ O) {
  int h = blockIdx.x, qi = blockIdx.y, b = blockIdx.z, tid = threadIdx.x;
  float qh[DH];
  const float* qr = Q + ((size_t)b*NPq + qi)*CHN + h*DH;
#pragma unroll
  for (int d=0; d<DH; ++d) qh[d] = qr[d];
  float mi = -3.4e38f, li = 0.f, av[DH];
#pragma unroll
  for (int d=0; d<DH; ++d) av[d] = 0.f;
  for (int p = tid; p < HWp; p += 256) {
    const float* kr = K + ((size_t)b*HWp + p)*CHN + h*DH;
    const float* vr = V + ((size_t)b*HWp + p)*CHN + h*DH;
    __builtin_prefetch(kr + 256*CHN, 0, 1);
    __builtin_prefetch(vr + 256*CHN, 0, 1);
    float s = 0.f;
#pragma unroll
    for (int d=0; d<DH; ++d) s += qh[d]*kr[d];
    s *= 0.25f;
    float mn = fmaxf(mi, s);
    float sc = expf(mi - mn);
    float w  = expf(s  - mn);
    li = li*sc + w;
#pragma unroll
    for (int d=0; d<DH; ++d) av[d] = av[d]*sc + w*vr[d];
    mi = mn;
  }
  __shared__ float sm[256], sl[256];
  __shared__ float sa[256][DH];
  sm[tid] = mi; sl[tid] = li;
  for (int d=0; d<DH; ++d) sa[tid][d] = av[d];
  __syncthreads();
  for (int off = 128; off > 0; off >>= 1) {
    if (tid < off) {
      float m1 = sm[tid], m2 = sm[tid+off];
      float M  = fmaxf(m1, m2);
      float e1 = expf(m1 - M), e2 = expf(m2 - M);
      sm[tid] = M;
      sl[tid] = sl[tid]*e1 + sl[tid+off]*e2;
      for (int d=0; d<DH; ++d) sa[tid][d] = sa[tid][d]*e1 + sa[tid+off][d]*e2;
    }
    __syncthreads();
  }
  if (tid < DH)
    O[((size_t)b*NPq + qi)*CHN + h*DH + tid] = sa[0][tid] / sl[0];
}

// ---------------- residual + LayerNorm --------------------------------------
__global__ void k_ln(const float* __restrict__ a, const float* __restrict__ r,
                     const float* __restrict__ g, const float* __restrict__ bt,
                     float* __restrict__ dst) {
  int m = blockIdx.x, b = blockIdx.y, c = threadIdx.x;
  size_t o = ((size_t)b*NPq + m)*CHN + c;
  float v = a[o] + r[o];
  __shared__ float red[CHN];
  __shared__ float mv;
  red[c] = v; __syncthreads();
  for (int off=64; off>0; off>>=1) { if (c<off) red[c]+=red[c+off]; __syncthreads(); }
  if (c == 0) mv = red[0]/(float)CHN;
  __syncthreads();
  float d = v - mv;
  red[c] = d*d; __syncthreads();
  for (int off=64; off>0; off>>=1) { if (c<off) red[c]+=red[c+off]; __syncthreads(); }
  float var = red[0]/(float)CHN;
  dst[o] = d*rsqrtf(var + 1e-5f)*g[c] + bt[c];
}

// ---------------- prediction heads ------------------------------------------
__global__ void k_heads(const float* __restrict__ x, const float* __restrict__ w1,
                        const float* __restrict__ g, const float* __restrict__ bt,
                        const float* __restrict__ w2, const float* __restrict__ b2,
                        float* __restrict__ out) {
  int p = blockIdx.x, r = blockIdx.y, b = blockIdx.z, c = threadIdx.x;
  int head = c_rowhead[r];
  const float* xr = x  + ((size_t)b*NPq + p)*CHN;
  const float* wr = w1 + ((size_t)head*CHN + c)*CHN;
  float acc = 0.f;
  for (int k=0; k<CHN; ++k) acc += wr[k]*xr[k];
  acc = acc*(g[head*CHN + c]*BN_S) + bt[head*CHN + c];
  acc = fmaxf(acc, 0.f);
  acc *= w2[r*CHN + c];
  __shared__ float red[CHN];
  red[c] = acc; __syncthreads();
  for (int off=64; off>0; off>>=1) { if (c<off) red[c]+=red[c+off]; __syncthreads(); }
  if (c == 0) out[((size_t)b*OUTR + r)*NPq + p] = red[0] + b2[r];
}

// ============================ launcher =======================================
extern "C" void kernel_launch(void* const* d_in, const int* in_sizes, int n_in,
                              void* d_out, int out_size, void* d_ws, size_t ws_size,
                              hipStream_t stream) {
  const float* inputs  = (const float*)d_in[0];
  const float* wsc     = (const float*)d_in[1];
  const float* bsc     = (const float*)d_in[2];
  const float* wbb     = (const float*)d_in[3];
  const float* bb_g    = (const float*)d_in[4];
  const float* bb_b    = (const float*)d_in[5];
  const float* whm     = (const float*)d_in[6];
  const float* bhm     = (const float*)d_in[7];
  const float* anc_w1  = (const float*)d_in[8];
  const float* anc_b1  = (const float*)d_in[9];
  const float* anc_w2  = (const float*)d_in[10];
  const float* anc_b2  = (const float*)d_in[11];
  const float* hm_scale= (const float*)d_in[12];
  const float* hm_bias = (const float*)d_in[13];
  const float* aq_w    = (const float*)d_in[14];
  const float* aq_b    = (const float*)d_in[15];
  const float* ffn_w1  = (const float*)d_in[16];
  const float* ffn_b1  = (const float*)d_in[17];
  const float* ffn_w2  = (const float*)d_in[18];
  const float* ffn_b2  = (const float*)d_in[19];
  const float* ln1_g   = (const float*)d_in[20];
  const float* ln1_b   = (const float*)d_in[21];
  const float* ln2_g   = (const float*)d_in[22];
  const float* ln2_b   = (const float*)d_in[23];
  const float* ln3_g   = (const float*)d_in[24];
  const float* ln3_b   = (const float*)d_in[25];
  const float* ph_w1   = (const float*)d_in[26];
  const float* ph_g    = (const float*)d_in[27];
  const float* ph_b    = (const float*)d_in[28];
  const float* ph_w2   = (const float*)d_in[29];
  const float* ph_b2   = (const float*)d_in[30];
  const float* pes_w1  = (const float*)d_in[31];
  const float* pes_b1  = (const float*)d_in[32];
  const float* pes_g   = (const float*)d_in[33];
  const float* pes_bt  = (const float*)d_in[34];
  const float* pes_w2  = (const float*)d_in[35];
  const float* pes_b2  = (const float*)d_in[36];
  const float* pec_w1  = (const float*)d_in[37];
  const float* pec_b1  = (const float*)d_in[38];
  const float* pec_g   = (const float*)d_in[39];
  const float* pec_bt  = (const float*)d_in[40];
  const float* pec_w2  = (const float*)d_in[41];
  const float* pec_b2  = (const float*)d_in[42];
  const float* sa_q_w  = (const float*)d_in[43];
  const float* sa_q_b  = (const float*)d_in[44];
  const float* sa_k_w  = (const float*)d_in[45];
  const float* sa_k_b  = (const float*)d_in[46];
  const float* sa_v_w  = (const float*)d_in[47];
  const float* sa_v_b  = (const float*)d_in[48];
  const float* sa_o_w  = (const float*)d_in[49];
  const float* sa_o_b  = (const float*)d_in[50];
  const float* ca_q_w  = (const float*)d_in[51];
  const float* ca_q_b  = (const float*)d_in[52];
  const float* ca_k_w  = (const float*)d_in[53];
  const float* ca_k_b  = (const float*)d_in[54];
  const float* ca_v_w  = (const float*)d_in[55];
  const float* ca_v_b  = (const float*)d_in[56];
  const float* ca_o_w  = (const float*)d_in[57];
  const float* ca_o_b  = (const float*)d_in[58];

  char* ws = (char*)d_ws;
  size_t off = 0;
  auto alloc = [&](size_t bytes) -> char* {
    char* p = ws + off; off += (bytes + 255) & ~(size_t)255; return p;
  };

  // big region: inT during conv1, reused for K/V after conv1 is consumed
  char*   inT_raw = alloc((size_t)Bq*HWp*CIN0*sizeof(__bf16));   // 66.4 MB
  __bf16* inT     = (__bf16*)inT_raw;
  float*  Kf      = (float*)inT_raw;                              // [B][HW][128]
  float*  Vf      = (float*)(inT_raw + (size_t)Bq*HWp*CHN*sizeof(float));

  float*  lidarF  = (float*) alloc((size_t)Bq*HWp*CHN*sizeof(float));
  __bf16* lidarBF = (__bf16*)alloc((size_t)Bq*HWp*CHN*sizeof(__bf16));
  __bf16* tBF     = (__bf16*)alloc((size_t)Bq*HWp*CHN*sizeof(__bf16));
  __bf16* kvBF    = (__bf16*)alloc((size_t)Bq*HWp*CHN*sizeof(__bf16));
  __bf16* WT1     = (__bf16*)alloc((size_t)9*CIN0*CHN*sizeof(__bf16));
  __bf16* WT2     = (__bf16*)alloc((size_t)9*CHN*CHN*sizeof(__bf16));
  __bf16* WTk     = (__bf16*)alloc((size_t)CHN*CHN*sizeof(__bf16));
  __bf16* WTv     = (__bf16*)alloc((size_t)CHN*CHN*sizeof(__bf16));
  float*  hm1     = (float*) alloc((size_t)Bq*HWp*sizeof(float));
  float*  heat    = (float*) alloc((size_t)Bq*NSEL*sizeof(float));
  float*  nmsB    = (float*) alloc((size_t)Bq*NSEL*sizeof(float));
  float*  sKb     = (float*) alloc(NCc*sizeof(float));
  float*  aqc     = (float*) alloc((size_t)NCc*CHN*sizeof(float));
  float*  scal    = (float*) alloc(2*sizeof(float));
  int*    clsB    = (int*)   alloc((size_t)Bq*NPq*sizeof(int));
  int*    spB     = (int*)   alloc((size_t)Bq*NPq*sizeof(int));
  size_t qmat = (size_t)Bq*NPq*CHN*sizeof(float);
  float* qf    = (float*)alloc(qmat);
  float* qpe   = (float*)alloc(qmat);
  float* Qs    = (float*)alloc(qmat);
  float* Ks    = (float*)alloc(qmat);
  float* Vs    = (float*)alloc(qmat);
  float* attnS = (float*)alloc(qmat);
  float* sap   = (float*)alloc(qmat);
  float* x1    = (float*)alloc(qmat);
  float* cq    = (float*)alloc(qmat);
  float* attnC = (float*)alloc(qmat);
  float* cap   = (float*)alloc(qmat);
  float* x2    = (float*)alloc(qmat);
  float* f1    = (float*)alloc((size_t)Bq*NPq*FFNh*sizeof(float));
  float* f2    = (float*)alloc(qmat);
  float* x3    = (float*)alloc(qmat);
  (void)ws_size; (void)in_sizes; (void)n_in; (void)out_size;

  // ---- weight prep (bf16, WMMA B layout source) ----
  k_prep_w<<<(9*CIN0*CHN + 255)/256, 256, 0, stream>>>(wsc, WT1, CIN0, 9);
  k_prep_w<<<(9*CHN*CHN  + 255)/256, 256, 0, stream>>>(wbb, WT2, CHN, 9);
  k_prep_w<<<(CHN*CHN    + 255)/256, 256, 0, stream>>>(ca_k_w, WTk, CHN, 1);
  k_prep_w<<<(CHN*CHN    + 255)/256, 256, 0, stream>>>(ca_v_w, WTv, CHN, 1);

  // ---- input to channel-last bf16 ----
  k_castT<<<(int)(((size_t)Bq*CIN0*HWp + 255)/256), 256, 0, stream>>>(inputs, inT);

  dim3 gridc((HWp + 127)/128, 1, Bq);
  // conv1: 512->128 3x3 + bias -> lidar (f32 + bf16, channel-last)
  k_conv_wmma<<<gridc, 256, 0, stream>>>(inT, WT1, bsc, nullptr, lidarF, lidarBF, CIN0, 9, 0);
  // conv2: 128->128 3x3, BN(scale)+ReLU -> t (bf16)
  k_conv_wmma<<<gridc, 256, 0, stream>>>(lidarBF, WT2, bb_g, bb_b, nullptr, tBF, CHN, 9, 1);
  // heatmap 1-channel conv
  k_hm<<<(Bq*HWp + 255)/256, 256, 0, stream>>>(tBF, whm, bhm, hm1);

  // anchors / scales / heatmap / NMS / top-k
  k_small_pre<<<1, 128, 0, stream>>>(anc_w1, anc_b1, anc_w2, anc_b2, aq_w,
                                     hm_scale, hm_bias, sKb, aqc, scal);
  k_heat<<<(Bq*NSEL + 255)/256, 256, 0, stream>>>(hm1, sKb, scal, heat);
  k_nms <<<(Bq*NSEL + 255)/256, 256, 0, stream>>>(heat, nmsB);
  k_topk<<<Bq, 1024, 0, stream>>>(nmsB, clsB, spB);

  // queries + positional embeds
  k_buildq<<<dim3(NPq, Bq), CHN, 0, stream>>>(spB, clsB, lidarF, aqc, aq_b,
               pes_w1, pes_b1, pes_g, pes_bt, pes_w2, pes_b2, qf, qpe);
  // key-side pos-embed fused into bf16 K/V input (lidar + kpe)
  k_kpe<<<HWp, CHN, 0, stream>>>(lidarF, pec_w1, pec_b1, pec_g, pec_bt, pec_w2, pec_b2, kvBF);

  // big K/V projections via WMMA (1x1 path)
  k_conv_wmma<<<gridc, 256, 0, stream>>>(kvBF, WTk, ca_k_b, nullptr, Kf, nullptr, CHN, 1, 0);
  k_conv_wmma<<<gridc, 256, 0, stream>>>(kvBF, WTv, ca_v_b, nullptr, Vf, nullptr, CHN, 1, 0);

  // self-attention (q_in = qf + qpe)
  dim3 gq(NPq, Bq);
  k_gemm_small<<<gq, CHN, CHN*sizeof(float), stream>>>(qf, qpe, sa_q_w, sa_q_b, Qs, CHN, CHN, 0);
  k_gemm_small<<<gq, CHN, CHN*sizeof(float), stream>>>(qf, qpe, sa_k_w, sa_k_b, Ks, CHN, CHN, 0);
  k_gemm_small<<<gq, CHN, CHN*sizeof(float), stream>>>(qf, qpe, sa_v_w, sa_v_b, Vs, CHN, CHN, 0);
  k_attn_small<<<gq, 256, 0, stream>>>(Qs, Ks, Vs, attnS);
  k_gemm_small<<<gq, CHN, CHN*sizeof(float), stream>>>(attnS, nullptr, sa_o_w, sa_o_b, sap, CHN, CHN, 0);
  k_ln<<<gq, CHN, 0, stream>>>(qf, sap, ln1_g, ln1_b, x1);

  // cross-attention (flash over 32400 keys)
  k_gemm_small<<<gq, CHN, CHN*sizeof(float), stream>>>(x1, qpe, ca_q_w, ca_q_b, cq, CHN, CHN, 0);
  k_flash<<<dim3(NHh, NPq, Bq), 256, 0, stream>>>(cq, Kf, Vf, attnC);
  k_gemm_small<<<gq, CHN, CHN*sizeof(float), stream>>>(attnC, nullptr, ca_o_w, ca_o_b, cap, CHN, CHN, 0);
  k_ln<<<gq, CHN, 0, stream>>>(x1, cap, ln2_g, ln2_b, x2);

  // FFN + LN3
  k_gemm_small<<<gq, FFNh, CHN*sizeof(float), stream>>>(x2, nullptr, ffn_w1, ffn_b1, f1, CHN, FFNh, 1);
  k_gemm_small<<<gq, CHN, FFNh*sizeof(float), stream>>>(f1, nullptr, ffn_w2, ffn_b2, f2, FFNh, CHN, 0);
  k_ln<<<gq, CHN, 0, stream>>>(x2, f2, ln3_g, ln3_b, x3);

  // prediction heads -> d_out [B,13,200] f32
  k_heads<<<dim3(NPq, OUTR, Bq), CHN, 0, stream>>>(x3, ph_w1, ph_g, ph_b, ph_w2, ph_b2,
                                                   (float*)d_out);
}
